// MllaBlock_37984690766444
// MI455X (gfx1250) — compile-verified, matrix operand order checked by reading, not run on hardware
//
#include <hip/hip_runtime.h>
#include <hip/hip_bf16.h>
#include <math.h>

// ---------- types for WMMA ----------
typedef _Float16 v8h  __attribute__((ext_vector_type(8)));
typedef _Float16 v16h __attribute__((ext_vector_type(16)));
typedef float    v8f  __attribute__((ext_vector_type(8)));

#define L_TOK   4096
#define C_DIM   768
#define HEADS   12
#define HD      64
#define HID     3072
#define LDT     40   // LDS row stride in halves (80B, 16B aligned)

// CDNA5 async global->LDS copy path (ASYNCcnt-tracked), with safe fallback.
#if defined(__HIP_DEVICE_COMPILE__) && defined(__gfx1250__) && \
    __has_builtin(__builtin_amdgcn_global_load_async_to_lds_b128)
#define ASYNC_COPY 1
#else
#define ASYNC_COPY 0
#endif

#if ASYNC_COPY
typedef int v4i __attribute__((vector_size(4 * sizeof(int))));
typedef __attribute__((address_space(1))) v4i* g_v4i_p;   // global src
typedef __attribute__((address_space(3))) v4i* l_v4i_p;   // LDS dst
__device__ __forceinline__ void async_b128(const void* g, void* l) {
    __builtin_amdgcn_global_load_async_to_lds_b128((g_v4i_p)g, (l_v4i_p)l, 0, 0);
}
__device__ __forceinline__ void wait_async0() {
#if __has_builtin(__builtin_amdgcn_s_wait_asynccnt)
    __builtin_amdgcn_s_wait_asynccnt(0);
#else
    asm volatile("s_wait_asynccnt 0x0" ::: "memory");
#endif
}
#endif

// Gather one 16x32-f16 WMMA operand fragment for this lane from an LDS tile
// row (row-major, stride LDT halves). Per ISA 7.12.2: lane holds k = half*8..+7
// and k = 16+half*8..+7  ->  two contiguous 16-byte LDS loads.
__device__ __forceinline__ v16h ld_frag(const _Float16* rowp, int half8) {
    v8h lo = *(const v8h*)(rowp + half8);
    v8h hi = *(const v8h*)(rowp + 16 + half8);
    return __builtin_shufflevector(lo, hi, 0,1,2,3,4,5,6,7,8,9,10,11,12,13,14,15);
}

__device__ __forceinline__ v8f wmma_f16(v16h a, v16h b, v8f c) {
    return __builtin_amdgcn_wmma_f32_16x16x32_f16(false, a, false, b,
                                                  (short)0, c, false, false);
}

// ======================================================================
// Generic WMMA GEMM on pre-converted f16 operands:
//   C[M,N] = epi( A[M,K] @ W[N,K]^T + bias )
// Block tile 128x128, BK=32, 256 thr = 8 waves (2M x 4N), 8 WMMA/wave/step.
// EPI: 0 = bias, 1 = bias+SiLU, 2 = bias+residual, 3 = bias+elu+1.
// Staging: double-buffered async-to-LDS on CDNA5 (overlap tile k+1 loads
// with tile-k WMMAs, s_wait_asynccnt before barrier); b128 ld/st fallback.
// ======================================================================
template<int EPI, typename OutT>
__global__ __launch_bounds__(256) void gemm_wmma(
        const _Float16* __restrict__ A, int lda,
        const _Float16* __restrict__ W,          // (Ntot, K) row-major
        const float* __restrict__ bias,
        const float* __restrict__ resid,
        OutT* __restrict__ C, int ldc, int K)
{
    __shared__ __align__(16) _Float16 As[2 * 128 * LDT];
    __shared__ __align__(16) _Float16 Bs[2 * 128 * LDT];

    const int tid  = threadIdx.x;
    const int lane = tid & 31;
    const int wid  = tid >> 5;
    const int wm   = wid >> 2;          // 0..1
    const int wn   = wid & 3;           // 0..3
    const int m0   = blockIdx.y * 128;
    const int n0   = blockIdx.x * 128;

    v8f acc[4][2] = {};

    const int half8 = (lane >> 4) * 8;
    const int r16   = lane & 15;

#if ASYNC_COPY
    auto stage = [&](int k0, int buf) {
        _Float16* Asb = As + buf * 128 * LDT;
        _Float16* Bsb = Bs + buf * 128 * LDT;
        for (int t = tid; t < 128 * 4; t += 256) {
            const int i = t >> 2, j8 = (t & 3) * 8;
            async_b128(&A[(size_t)(m0 + i) * lda + k0 + j8], &Asb[i * LDT + j8]);
            async_b128(&W[(size_t)(n0 + i) * K   + k0 + j8], &Bsb[i * LDT + j8]);
        }
    };
    stage(0, 0);
    wait_async0();
    __syncthreads();
    int cur = 0;
    for (int k0 = 0; k0 < K; k0 += 32) {
        const bool more = (k0 + 32) < K;
        if (more) stage(k0 + 32, cur ^ 1);          // overlap with compute

        const _Float16* Asb = As + cur * 128 * LDT;
        const _Float16* Bsb = Bs + cur * 128 * LDT;
        v16h bf[2];
        #pragma unroll
        for (int ni = 0; ni < 2; ++ni)
            bf[ni] = ld_frag(&Bsb[(wn * 32 + ni * 16 + r16) * LDT], half8);
        #pragma unroll
        for (int mi = 0; mi < 4; ++mi) {
            v16h af = ld_frag(&Asb[(wm * 64 + mi * 16 + r16) * LDT], half8);
            #pragma unroll
            for (int ni = 0; ni < 2; ++ni)
                acc[mi][ni] = wmma_f16(af, bf[ni], acc[mi][ni]);
        }
        if (more) wait_async0();
        __syncthreads();
        cur ^= 1;
    }
#else
    for (int k0 = 0; k0 < K; k0 += 32) {
        for (int t = tid; t < 128 * 4; t += 256) {
            const int i = t >> 2, j8 = (t & 3) * 8;
            *(v8h*)&As[i * LDT + j8] = *(const v8h*)&A[(size_t)(m0 + i) * lda + k0 + j8];
            *(v8h*)&Bs[i * LDT + j8] = *(const v8h*)&W[(size_t)(n0 + i) * K   + k0 + j8];
        }
        __syncthreads();
        v16h bf[2];
        #pragma unroll
        for (int ni = 0; ni < 2; ++ni)
            bf[ni] = ld_frag(&Bs[(wn * 32 + ni * 16 + r16) * LDT], half8);
        #pragma unroll
        for (int mi = 0; mi < 4; ++mi) {
            v16h af = ld_frag(&As[(wm * 64 + mi * 16 + r16) * LDT], half8);
            #pragma unroll
            for (int ni = 0; ni < 2; ++ni)
                acc[mi][ni] = wmma_f16(af, bf[ni], acc[mi][ni]);
        }
        __syncthreads();
    }
#endif

    // D layout: row = (lane/16)*8 + e, col = lane%16 within each 16x16 tile
    const int rlo = (lane >> 4) * 8;
    #pragma unroll
    for (int mi = 0; mi < 4; ++mi)
        #pragma unroll
        for (int ni = 0; ni < 2; ++ni)
            #pragma unroll
            for (int e = 0; e < 8; ++e) {
                const int row = m0 + wm * 64 + mi * 16 + rlo + e;
                const int col = n0 + wn * 32 + ni * 16 + r16;
                float v = acc[mi][ni][e] + bias[col];
                if (EPI == 1) v = v / (1.f + __expf(-v));               // SiLU
                if (EPI == 2) v += resid[(size_t)row * ldc + col];      // +res
                if (EPI == 3) v = v > 0.f ? v + 1.f : __expf(v);        // elu+1
                C[(size_t)row * ldc + col] = (OutT)v;
            }
}

// ======================================================================
// kv[h] = (k_rope*s)^T @ (v*s) : per-head 64x64, K=4096.  One block per
// head, 4 waves; wave w owns M-subtile w, 4 N-subtiles.  A is the k-half of
// the f16 qk buffer read transposed while staging to LDS.
// ======================================================================
__global__ __launch_bounds__(128) void kv_wmma(
        const _Float16* __restrict__ qk16,  // (L, 2C): k_rope at col 768+
        const _Float16* __restrict__ v16,   // (L, C)
        _Float16* __restrict__ kvout)       // (H, 64, 64) f16
{
    __shared__ __align__(16) _Float16 As[64 * LDT];
    __shared__ __align__(16) _Float16 Bs[64 * LDT];

    const int h    = blockIdx.x;
    const int tid  = threadIdx.x;
    const int lane = tid & 31;
    const int wid  = tid >> 5;         // 0..3 -> M-subtile
    const int half8 = (lane >> 4) * 8;
    const int r16   = lane & 15;

    v8f acc[4] = {};

    for (int k0 = 0; k0 < L_TOK; k0 += 32) {
        for (int t = tid; t < 64 * 32; t += 128) {
            const int d = t & 63, j = t >> 6;      // transpose while staging
            As[d * LDT + j] = qk16[(size_t)(k0 + j) * (2 * C_DIM) + C_DIM + h * HD + d];
            Bs[d * LDT + j] = v16 [(size_t)(k0 + j) * C_DIM + h * HD + d];
        }
        __syncthreads();
        v16h af = ld_frag(&As[(wid * 16 + r16) * LDT], half8);
        #pragma unroll
        for (int ni = 0; ni < 4; ++ni) {
            v16h bf = ld_frag(&Bs[(ni * 16 + r16) * LDT], half8);
            acc[ni] = wmma_f16(af, bf, acc[ni]);
        }
        __syncthreads();
    }

    const int rlo = (lane >> 4) * 8;
    const float sc = 1.f / 4096.f;     // scale^2, scale = L^-0.5
    #pragma unroll
    for (int ni = 0; ni < 4; ++ni)
        #pragma unroll
        for (int e = 0; e < 8; ++e)
            kvout[(size_t)h * 4096 + (wid * 16 + rlo + e) * 64 + ni * 16 + r16] =
                (_Float16)(acc[ni][e] * sc);
}

// ======================================================================
// a[n, h*64+e] = z[h,n] * sum_d q_rope[n, h*64+d] * kv[h,d,e]
// Grid (Mtiles=32, heads=12); 8 waves, wave w owns M-subtile w x 64 cols.
// ======================================================================
__global__ __launch_bounds__(256) void attn_wmma(
        const _Float16* __restrict__ qk16,  // q_rope at col 0
        const _Float16* __restrict__ kv16,  // (H, 64, 64)
        const float* __restrict__ zb,       // (H, L)
        float* __restrict__ abuf)           // (L, C)
{
    __shared__ __align__(16) _Float16 As[128 * LDT];
    __shared__ __align__(16) _Float16 Bs[64 * LDT];

    const int h    = blockIdx.y;
    const int m0   = blockIdx.x * 128;
    const int tid  = threadIdx.x;
    const int lane = tid & 31;
    const int wid  = tid >> 5;         // 0..7 -> M-subtile
    const int half8 = (lane >> 4) * 8;
    const int r16   = lane & 15;

    v8f acc[4] = {};

    for (int kk = 0; kk < HD; kk += 32) {
        for (int t = tid; t < 128 * 4; t += 256) {   // A rows: 16B copies
            const int i = t >> 2, j8 = (t & 3) * 8;
#if ASYNC_COPY
            async_b128(&qk16[(size_t)(m0 + i) * (2 * C_DIM) + h * HD + kk + j8],
                       &As[i * LDT + j8]);
#else
            *(v8h*)&As[i * LDT + j8] =
                *(const v8h*)&qk16[(size_t)(m0 + i) * (2 * C_DIM) + h * HD + kk + j8];
#endif
        }
        for (int t = tid; t < 64 * 32; t += 256) {   // B: transpose-stage
            const int e = t & 63, j = t >> 6;
            Bs[e * LDT + j] = kv16[(size_t)h * 4096 + (kk + j) * 64 + e];
        }
#if ASYNC_COPY
        wait_async0();
#endif
        __syncthreads();
        v16h af = ld_frag(&As[(wid * 16 + r16) * LDT], half8);
        #pragma unroll
        for (int ni = 0; ni < 4; ++ni) {
            v16h bf = ld_frag(&Bs[(ni * 16 + r16) * LDT], half8);
            acc[ni] = wmma_f16(af, bf, acc[ni]);
        }
        __syncthreads();
    }

    const int rlo = (lane >> 4) * 8;
    #pragma unroll
    for (int ni = 0; ni < 4; ++ni)
        #pragma unroll
        for (int e = 0; e < 8; ++e) {
            const int row = m0 + wid * 16 + rlo + e;
            abuf[(size_t)row * C_DIM + h * HD + ni * 16 + r16] =
                acc[ni][e] * zb[(size_t)h * L_TOK + row];
        }
}

// ======================================================================
// Depthwise 3x3 conv on (C,64,64) stored token-major (L,C), fused adds:
// out = conv(src) + b + add0? + add1?
// ======================================================================
__global__ void dwconv_fuse(const float* __restrict__ src,
                            const float* __restrict__ w,
                            const float* __restrict__ b,
                            const float* __restrict__ add0,
                            const float* __restrict__ add1,
                            float* __restrict__ out)
{
    const int idx = blockIdx.x * blockDim.x + threadIdx.x;
    if (idx >= L_TOK * C_DIM) return;
    const int c = idx % C_DIM, n = idx / C_DIM;
    const int hh = n >> 6, ww = n & 63;
    float s = b[c];
    #pragma unroll
    for (int dh = -1; dh <= 1; ++dh) {
        const int h2 = hh + dh;
        if (h2 < 0 || h2 >= 64) continue;
        #pragma unroll
        for (int dw = -1; dw <= 1; ++dw) {
            const int w2 = ww + dw;
            if (w2 < 0 || w2 >= 64) continue;
            s += src[(size_t)(h2 * 64 + w2) * C_DIM + c] * w[c * 9 + (dh + 1) * 3 + (dw + 1)];
        }
    }
    if (add0) s += add0[idx];
    if (add1) s += add1[idx];
    out[idx] = s;
}

// ------- LayerNorm over C=768, one block/token; optional f16 copy -------
__global__ __launch_bounds__(256) void layernorm_k(
        const float* __restrict__ in, const float* __restrict__ w,
        const float* __restrict__ b, float* __restrict__ out,
        _Float16* __restrict__ out16)
{
    __shared__ float s1[256], s2[256];
    const int n = blockIdx.x, tid = threadIdx.x;
    const float* row = in + (size_t)n * C_DIM;
    const float a0 = row[tid], a1 = row[tid + 256], a2 = row[tid + 512];
    s1[tid] = a0 + a1 + a2;
    s2[tid] = a0 * a0 + a1 * a1 + a2 * a2;
    __syncthreads();
    for (int s = 128; s > 0; s >>= 1) {
        if (tid < s) { s1[tid] += s1[tid + s]; s2[tid] += s2[tid + s]; }
        __syncthreads();
    }
    const float mu  = s1[0] * (1.f / C_DIM);
    const float var = s2[0] * (1.f / C_DIM) - mu * mu;
    const float inv = rsqrtf(var + 1e-5f);
    const float r0 = (a0 - mu) * inv * w[tid]       + b[tid];
    const float r1 = (a1 - mu) * inv * w[tid + 256] + b[tid + 256];
    const float r2 = (a2 - mu) * inv * w[tid + 512] + b[tid + 512];
    float* o = out + (size_t)n * C_DIM;
    o[tid] = r0; o[tid + 256] = r1; o[tid + 512] = r2;
    if (out16) {
        _Float16* o16 = out16 + (size_t)n * C_DIM;
        o16[tid] = (_Float16)r0;
        o16[tid + 256] = (_Float16)r1;
        o16[tid + 512] = (_Float16)r2;
    }
}

// ---------------- f32 -> f16 conversion (weights) ----------------
__global__ void cvt16_k(const float* __restrict__ in, _Float16* __restrict__ out, int n)
{
    const int i = blockIdx.x * blockDim.x + threadIdx.x;
    if (i < n) out[i] = (_Float16)in[i];
}

// ---------------- k.mean over tokens -> (C,) ----------------
__global__ __launch_bounds__(256) void kmean_k(const float* __restrict__ qk,
                                               float* __restrict__ km)
{
    __shared__ float s[256];
    const int cc = blockIdx.x, tid = threadIdx.x;
    float acc = 0.f;
    for (int n = tid; n < L_TOK; n += 256)
        acc += qk[(size_t)n * (2 * C_DIM) + C_DIM + cc];
    s[tid] = acc;
    __syncthreads();
    for (int st = 128; st > 0; st >>= 1) {
        if (tid < st) s[tid] += s[tid + st];
        __syncthreads();
    }
    if (tid == 0) km[cc] = s[0] * (1.f / L_TOK);
}

// ---------------- z[h,n] = 1/(q[n,h,:].k_mean[h,:] + 1e-6) ----------------
__global__ void z_k(const float* __restrict__ qk, const float* __restrict__ km,
                    float* __restrict__ z)
{
    const int idx = blockIdx.x * blockDim.x + threadIdx.x;
    if (idx >= HEADS * L_TOK) return;
    const int h = idx >> 12, n = idx & 4095;
    float s = 0.f;
    #pragma unroll 8
    for (int d = 0; d < HD; ++d)
        s += qk[(size_t)n * (2 * C_DIM) + h * HD + d] * km[h * HD + d];
    z[idx] = 1.f / (s + 1e-6f);
}

// ------- axial 2D RoPE on q and k; reads f32, writes rotated f16 -------
__global__ void rope_k(const float* __restrict__ qk, _Float16* __restrict__ qk16)
{
    const int idx = blockIdx.x * blockDim.x + threadIdx.x;
    if (idx >= L_TOK * (C_DIM / 2)) return;
    const int n = idx / (C_DIM / 2), p = idx % (C_DIM / 2);
    const int hp = n >> 6, wp = n & 63;
    const int i  = (p < 192) ? p : p - 192;
    const float pos = (float)((p < 192) ? hp : wp);
    const float theta = __expf(-(float)i * (9.210340371976184f / 192.f)); // ln(1e4)
    const float ang = pos * theta;
    const float cs = __cosf(ang), sn = __sinf(ang);
    size_t bq = (size_t)n * (2 * C_DIM) + 2 * p;
    float x0 = qk[bq], x1 = qk[bq + 1];
    qk16[bq]     = (_Float16)(x0 * cs - x1 * sn);
    qk16[bq + 1] = (_Float16)(x0 * sn + x1 * cs);
    const size_t bk = bq + C_DIM;
    x0 = qk[bk]; x1 = qk[bk + 1];
    qk16[bk]     = (_Float16)(x0 * cs - x1 * sn);
    qk16[bk + 1] = (_Float16)(x0 * sn + x1 * cs);
}

// ======================================================================
extern "C" void kernel_launch(void* const* d_in, const int* in_sizes, int n_in,
                              void* d_out, int out_size, void* d_ws, size_t ws_size,
                              hipStream_t stream)
{
    (void)in_sizes; (void)n_in; (void)out_size; (void)ws_size;
    const float* x       = (const float*)d_in[0];
    const float* cpe1_w  = (const float*)d_in[1];
    const float* cpe1_b  = (const float*)d_in[2];
    const float* norm1_w = (const float*)d_in[3];
    const float* norm1_b = (const float*)d_in[4];
    const float* qk_w    = (const float*)d_in[5];
    const float* qk_b    = (const float*)d_in[6];
    const float* lepe_w  = (const float*)d_in[7];
    const float* lepe_b  = (const float*)d_in[8];
    const float* cpe2_w  = (const float*)d_in[9];
    const float* cpe2_b  = (const float*)d_in[10];
    const float* norm2_w = (const float*)d_in[11];
    const float* norm2_b = (const float*)d_in[12];
    const float* fc1_w   = (const float*)d_in[13];
    const float* fc1_b   = (const float*)d_in[14];
    const float* fc2_w   = (const float*)d_in[15];
    const float* fc2_b   = (const float*)d_in[16];

    float* ws = (float*)d_ws;
    const size_t LC = (size_t)L_TOK * C_DIM;

    // f32 scratch
    float* xr     = ws;             // x + cpe1
    float* x1n    = ws + 1 * LC;    // LN1 output (= v)
    float* qkbuf  = ws + 2 * LC;    // (L, 2C): elu(q)+1 | elu(k)+1 (pre-RoPE)
    float* abuf   = ws + 4 * LC;    // attention out
    float* x2     = ws + 5 * LC;    // x + a + lepe
    float* x3     = ws + 6 * LC;    // x2 + cpe2
    float* ybuf   = ws + 7 * LC;    // LN2 output
    float* km     = ws + 8 * LC;            // 768
    float* zb     = km + 1024;              // 12*4096

    // f16 scratch
    _Float16* h16 = (_Float16*)(ws + 8 * LC + 65536);
    _Float16* x1n16   = h16;                    // (L, C)
    _Float16* qk16    = x1n16 + LC;             // (L, 2C) post-RoPE q|k
    _Float16* y16     = qk16 + 2 * LC;          // (L, C)
    _Float16* hid16   = y16 + LC;               // (L, 4C)
    _Float16* kv16    = hid16 + 4 * LC;         // (H, 64, 64)
    _Float16* wqk16   = kv16 + (size_t)HEADS * 4096;        // (1536, 768)
    _Float16* wfc1_16 = wqk16 + (size_t)2 * C_DIM * C_DIM;  // (3072, 768)
    _Float16* wfc2_16 = wfc1_16 + (size_t)HID * C_DIM;      // (768, 3072)

    const int NE = L_TOK * C_DIM;
    const dim3 B(256);

    // one-time (per launch) weight conversion to f16
    cvt16_k<<<(2 * C_DIM * C_DIM + 255) / 256, B, 0, stream>>>(qk_w,  wqk16,   2 * C_DIM * C_DIM);
    cvt16_k<<<(HID * C_DIM + 255) / 256,     B, 0, stream>>>(fc1_w, wfc1_16, HID * C_DIM);
    cvt16_k<<<(HID * C_DIM + 255) / 256,     B, 0, stream>>>(fc2_w, wfc2_16, HID * C_DIM);

    // CPE1 + residual, LN1 (emits f32 + f16)
    dwconv_fuse<<<(NE + 255) / 256, B, 0, stream>>>(x, cpe1_w, cpe1_b, x, nullptr, xr);
    layernorm_k<<<L_TOK, B, 0, stream>>>(xr, norm1_w, norm1_b, x1n, x1n16);

    // qk = elu(x1n @ qk_w^T + qk_b)+1  (M=4096, N=1536, K=768; EPI=3)
    gemm_wmma<3, float><<<dim3(1536 / 128, L_TOK / 128), B, 0, stream>>>(
        x1n16, C_DIM, wqk16, qk_b, nullptr, qkbuf, 2 * C_DIM, C_DIM);

    // k-mean, z, then RoPE -> f16 qk buffer
    kmean_k<<<C_DIM, B, 0, stream>>>(qkbuf, km);
    z_k<<<(HEADS * L_TOK + 255) / 256, B, 0, stream>>>(qkbuf, km, zb);
    rope_k<<<(L_TOK * (C_DIM / 2) + 255) / 256, B, 0, stream>>>(qkbuf, qk16);

    // kv = (k_rope*s)^T @ (v*s), a = z * (q_rope @ kv)
    kv_wmma<<<HEADS, dim3(128), 0, stream>>>(qk16, x1n16, kv16);
    attn_wmma<<<dim3(L_TOK / 128, HEADS), B, 0, stream>>>(qk16, kv16, zb, abuf);

    // x2 = x + a + lepe(x1n);  x3 = x2 + cpe2(x2);  y = LN2(x3) (+f16)
    dwconv_fuse<<<(NE + 255) / 256, B, 0, stream>>>(x1n, lepe_w, lepe_b, x, abuf, x2);
    dwconv_fuse<<<(NE + 255) / 256, B, 0, stream>>>(x2, cpe2_w, cpe2_b, x2, nullptr, x3);
    layernorm_k<<<L_TOK, B, 0, stream>>>(x3, norm2_w, norm2_b, ybuf, y16);

    // MLP: hidden(f16) = silu(y @ fc1^T + b1); out = x3 + hidden @ fc2^T + b2
    gemm_wmma<1, _Float16><<<dim3(HID / 128, L_TOK / 128), B, 0, stream>>>(
        y16, C_DIM, wfc1_16, fc1_b, nullptr, hid16, HID, C_DIM);
    gemm_wmma<2, float><<<dim3(C_DIM / 128, L_TOK / 128), B, 0, stream>>>(
        hid16, HID, wfc2_16, fc2_b, x3, (float*)d_out, C_DIM, HID);
}
